// Decoder_2688649527663
// MI455X (gfx1250) — matrix operand order repaired and checked by reading, not compiled, not run on hardware
//
#include <hip/hip_runtime.h>

// ---------- types ----------
typedef __attribute__((ext_vector_type(8)))  __bf16 v8bf;
typedef __attribute__((ext_vector_type(16))) __bf16 v16bf;
typedef __attribute__((ext_vector_type(8)))  float  v8f;

#define BATCH 32
#define VOCAB 32000
#define EMBD  512
#define ZDIM  512
#define HDIM  1024
#define KG    2048   // EMB+ZD+H  (gates GEMM K)
#define FOURH 4096

// ---------- helpers ----------
__device__ __forceinline__ __bf16 f2bf(float f) {
    union { float f; unsigned u; } x; x.f = f;
    unsigned r = x.u + 0x7FFFu + ((x.u >> 16) & 1u);   // round-to-nearest-even
    union { unsigned short s; __bf16 b; } y; y.s = (unsigned short)(r >> 16);
    return y.b;
}

__device__ __forceinline__ float sigmoidf_(float x) { return 1.0f / (1.0f + expf(-x)); }

__device__ __forceinline__ unsigned hash3(unsigned a, unsigned b, unsigned c) {
    unsigned h = a * 0x9E3779B1u ^ b * 0x85EBCA77u ^ c * 0xC2B2AE3Du;
    h ^= h >> 16; h *= 0x7FEB352Du;
    h ^= h >> 15; h *= 0x846CA68Bu;
    h ^= h >> 16;
    return h;
}

// A fragment (16x32 bf16, M x K), contiguous k order (same as load_b_frag):
// lanes 0-15: row = lane, K {kb..kb+15}
// lanes16-31: row = lane-16, K {kb+16..kb+31}
__device__ __forceinline__ v16bf load_a_frag(const __bf16* __restrict__ base,
                                             int ldk, int kk, int lane) {
    int m    = lane & 15;
    int half = lane >> 4;
    int kb   = kk * 32 + half * 16;
    const __bf16* p = base + m * ldk + kb;
    v8bf lo = *(const v8bf*)p;
    v8bf hi = *(const v8bf*)(p + 8);
    v16bf r;
#pragma unroll
    for (int i = 0; i < 8; ++i) { r[i] = lo[i]; r[i + 8] = hi[i]; }
    return r;
}

// B fragment (32x16 bf16, K x N). Weights stored row-major [N, K], so
// element (k, n) = W[n*ldk + k]. lanes 0-15: K 0..15 ; lanes 16-31: K 16..31.
__device__ __forceinline__ v16bf load_b_frag(const __bf16* __restrict__ ncol_base,
                                             int ldk, int kk, int lane) {
    int n    = lane & 15;
    int half = lane >> 4;
    int kb   = kk * 32 + half * 16;
    return *(const v16bf*)(ncol_base + n * ldk + kb);
}

// ---------- one-time conversion kernels ----------
__global__ __launch_bounds__(256) void cvt_gate_kernel(const float* __restrict__ w_ih,
                                                       const float* __restrict__ w_hh,
                                                       __bf16* __restrict__ Wg) {
    int idx = blockIdx.x * 256 + threadIdx.x;          // < 4096*2048
    int n = idx >> 11, k = idx & 2047;
    float v = (k < 1024) ? w_ih[n * 1024 + k] : w_hh[n * 1024 + (k - 1024)];
    Wg[idx] = f2bf(v);
}

__global__ __launch_bounds__(256) void cvt_out_kernel(const float* __restrict__ w_out,
                                                      __bf16* __restrict__ Wout) {
    unsigned idx = blockIdx.x * 256u + threadIdx.x;
    if (idx < (unsigned)VOCAB * HDIM) Wout[idx] = f2bf(w_out[idx]);
}

__global__ __launch_bounds__(256) void init_kernel(float* __restrict__ Cst,
                                                   __bf16* __restrict__ Hbf,
                                                   int* __restrict__ s_prev,
                                                   const int* __restrict__ start_id) {
    int idx = blockIdx.x * 256 + threadIdx.x;
    if (idx < BATCH * HDIM) { Cst[idx] = 0.0f; Hbf[idx] = f2bf(0.0f); }
    if (idx < BATCH) s_prev[idx] = start_id[0];
}

// ---------- per-step: build A = [emb[s] | z | h] as bf16 [32, 2048] ----------
__global__ __launch_bounds__(256) void prepare_A_kernel(const float* __restrict__ z,
                                                        const float* __restrict__ emb,
                                                        const int* __restrict__ s_prev,
                                                        const __bf16* __restrict__ Hbf,
                                                        __bf16* __restrict__ A) {
    int idx = blockIdx.x * 256 + threadIdx.x;          // < 32*2048
    int m = idx >> 11, k = idx & 2047;
    __bf16 v;
    if (k < EMBD)       v = f2bf(emb[(size_t)s_prev[m] * EMBD + k]);
    else if (k < 1024)  v = f2bf(z[m * ZDIM + (k - EMBD)]);
    else                v = Hbf[m * HDIM + (k - 1024)];
    A[idx] = v;
}

// ---------- per-step: gates GEMM + fused LSTM cell ----------
// 64 waves total; wave w owns hidden cols [w*16, w*16+16) across all 4 gates,
// so i/f/g/o for one (row, hidden) element land in the same lane/register slot
// and the cell nonlinearity fuses entirely in-register.
// Wg (16.8 MB bf16) is L2-resident across all 100 steps -> no prefetch needed.
__global__ __launch_bounds__(128) void gates_cell_kernel(const __bf16* __restrict__ A,    // [32,2048]
                                                         const __bf16* __restrict__ Wg,   // [4096,2048]
                                                         const float* __restrict__ bias,  // [4096]
                                                         float* __restrict__ Cst,         // [32,1024]
                                                         __bf16* __restrict__ Hbf) {      // [32,1024]
    const int lane = threadIdx.x & 31;
    const int wave = blockIdx.x * (blockDim.x >> 5) + (threadIdx.x >> 5);  // 0..63
    const int nb   = wave * 16;

    v8f acc[8] = {};  // [gate*2 + mtile]

    for (int kk = 0; kk < KG / 32; ++kk) {
        v16bf a0 = load_a_frag(A,           KG, kk, lane);
        v16bf a1 = load_a_frag(A + 16 * KG, KG, kk, lane);
#pragma unroll
        for (int g = 0; g < 4; ++g) {
            const __bf16* bcol = Wg + (size_t)(g * HDIM + nb) * KG;
            v16bf bm = load_b_frag(bcol, KG, kk, lane);
            acc[g * 2 + 0] = __builtin_amdgcn_wmma_f32_16x16x32_bf16(
                false, a0, false, bm, (short)0, acc[g * 2 + 0], false, false);
            acc[g * 2 + 1] = __builtin_amdgcn_wmma_f32_16x16x32_bf16(
                false, a1, false, bm, (short)0, acc[g * 2 + 1], false, false);
        }
    }

    const int n = lane & 15, half = lane >> 4;
    const int col = nb + n;
    const float bi = bias[col], bf_ = bias[HDIM + col],
                bg = bias[2 * HDIM + col], bo = bias[3 * HDIM + col];
#pragma unroll
    for (int mt = 0; mt < 2; ++mt) {
#pragma unroll
        for (int r = 0; r < 8; ++r) {
            int m = mt * 16 + r + half * 8;  // C/D layout: VGPR r -> row r (+8 upper half)
            float iv = sigmoidf_(acc[0 * 2 + mt][r] + bi);
            float fv = sigmoidf_(acc[1 * 2 + mt][r] + bf_);
            float gv = tanhf    (acc[2 * 2 + mt][r] + bg);
            float ov = sigmoidf_(acc[3 * 2 + mt][r] + bo);
            float c  = fv * Cst[m * HDIM + col] + iv * gv;
            Cst[m * HDIM + col] = c;
            Hbf[m * HDIM + col] = f2bf(ov * tanhf(c));
        }
    }
}

// ---------- per-step: logits GEMM [32,1024] x [1024,32000] ----------
// w_out is the dominant 65 MB/step stream (L2-resident after step 0).
// Unconditional WGP-scope prefetch 8 K-blocks ahead hides L2->L0 latency
// without any branches in the loop. Tail over-run stays inside the workspace.
__global__ __launch_bounds__(256) void logits_kernel(const __bf16* __restrict__ Hbf,   // [32,1024]
                                                     const __bf16* __restrict__ Wout,  // [32000,1024]
                                                     const float* __restrict__ b_out,  // [32000]
                                                     float* __restrict__ logits) {     // [32,32000]
    const int lane = threadIdx.x & 31;
    const int wave = blockIdx.x * (blockDim.x >> 5) + (threadIdx.x >> 5);  // 0..1999
    const int nb   = wave * 16;
    const __bf16* bcol = Wout + (size_t)nb * HDIM;
    const __bf16* pf   = bcol + (lane & 15) * HDIM;

    v8f acc0 = {}, acc1 = {};
    for (int kk = 0; kk < HDIM / 32; ++kk) {
        __builtin_prefetch(pf + kk * 32 + 256, 0, 3);   // WGP scope, 8 blocks ahead
        v16bf a0 = load_a_frag(Hbf,             HDIM, kk, lane);
        v16bf a1 = load_a_frag(Hbf + 16 * HDIM, HDIM, kk, lane);
        v16bf bm = load_b_frag(bcol, HDIM, kk, lane);
        acc0 = __builtin_amdgcn_wmma_f32_16x16x32_bf16(false, a0, false, bm, (short)0, acc0, false, false);
        acc1 = __builtin_amdgcn_wmma_f32_16x16x32_bf16(false, a1, false, bm, (short)0, acc1, false, false);
    }

    const int n = lane & 15, half = lane >> 4;
    const int col = nb + n;
    const float bb = b_out[col];
#pragma unroll
    for (int r = 0; r < 8; ++r) {
        logits[(r + half * 8) * VOCAB + col]      = acc0[r] + bb;
        logits[(16 + r + half * 8) * VOCAB + col] = acc1[r] + bb;
    }
}

// ---------- per-step: softmax + Gumbel-max sampling ----------
__global__ __launch_bounds__(256) void softmax_sample_kernel(const float* __restrict__ logits, // [32,V]
                                                             float* __restrict__ out,          // d_out
                                                             int* __restrict__ s_prev,
                                                             int t, int T) {
    __shared__ float red[256];
    __shared__ int   redi[256];
    const int b = blockIdx.x, tid = threadIdx.x;
    const float* row = logits + (size_t)b * VOCAB;

    // pass 1: row max
    float mx = -3.4e38f;
    for (int v = tid; v < VOCAB; v += 256) mx = fmaxf(mx, row[v]);
    red[tid] = mx; __syncthreads();
    for (int s = 128; s > 0; s >>= 1) {
        if (tid < s) red[tid] = fmaxf(red[tid], red[tid + s]);
        __syncthreads();
    }
    mx = red[0]; __syncthreads();

    // pass 2: sum of exp + Gumbel argmax
    float sum = 0.0f, best = -3.4e38f; int bidx = 0;
    for (int v = tid; v < VOCAB; v += 256) {
        float l = row[v];
        sum += expf(l - mx);
        unsigned h = hash3(0x5A17u + (unsigned)t, (unsigned)b, (unsigned)v);
        float u = (float)(h >> 8) * (1.0f / 16777216.0f) + 5.9604645e-8f;
        float gl = l - logf(-logf(u));
        if (gl > best) { best = gl; bidx = v; }
    }
    red[tid] = sum; __syncthreads();
    for (int s = 128; s > 0; s >>= 1) {
        if (tid < s) red[tid] += red[tid + s];
        __syncthreads();
    }
    float tot = red[0]; __syncthreads();

    red[tid] = best; redi[tid] = bidx; __syncthreads();
    for (int s = 128; s > 0; s >>= 1) {
        if (tid < s && red[tid + s] > red[tid]) { red[tid] = red[tid + s]; redi[tid] = redi[tid + s]; }
        __syncthreads();
    }
    if (tid == 0) {
        s_prev[b] = redi[0];
        out[(size_t)b * T + t] = (float)redi[0];   // s_ts section [B,T]
    }

    // pass 3: write probabilities p_t  (section [B,T,V] after [B,T])
    float inv = 1.0f / tot;
    float* pout = out + (size_t)BATCH * T + ((size_t)b * T + t) * (size_t)VOCAB;
    for (int v = tid; v < VOCAB; v += 256) pout[v] = expf(row[v] - mx) * inv;
}

// ---------- launcher ----------
extern "C" void kernel_launch(void* const* d_in, const int* in_sizes, int n_in,
                              void* d_out, int out_size, void* d_ws, size_t ws_size,
                              hipStream_t stream) {
    const float* z        = (const float*)d_in[0];
    const float* emb      = (const float*)d_in[1];
    const float* w_ih     = (const float*)d_in[2];
    const float* w_hh     = (const float*)d_in[3];
    const float* bias     = (const float*)d_in[4];
    const float* w_out    = (const float*)d_in[5];
    const float* b_out    = (const float*)d_in[6];
    const int*   start_id = (const int*)d_in[7];
    float* out = (float*)d_out;

    // workspace carve-out (256B aligned)
    char* ws = (char*)d_ws;
    size_t off = 0;
    auto carve = [&](size_t bytes) -> void* {
        void* p = ws + off;
        off += (bytes + 255) & ~(size_t)255;
        return p;
    };
    __bf16* Wg     = (__bf16*)carve((size_t)FOURH * KG * 2);       // 16.8 MB
    __bf16* Wout   = (__bf16*)carve((size_t)VOCAB * HDIM * 2);     // 65.5 MB
    __bf16* A      = (__bf16*)carve((size_t)BATCH * KG * 2);
    __bf16* Hbf    = (__bf16*)carve((size_t)BATCH * HDIM * 2);
    float*  Cst    = (float*) carve((size_t)BATCH * HDIM * 4);
    float*  logits = (float*) carve((size_t)BATCH * VOCAB * 4);    // 4.1 MB
    int*    s_prev = (int*)   carve(BATCH * 4);
    (void)ws_size; (void)n_in; (void)in_sizes;

    int T = out_size / (BATCH * (1 + VOCAB));
    if (T <= 0) T = 100;

    cvt_gate_kernel<<<(FOURH * KG) / 256, 256, 0, stream>>>(w_ih, w_hh, Wg);
    cvt_out_kernel<<<((size_t)VOCAB * HDIM + 255) / 256, 256, 0, stream>>>(w_out, Wout);
    init_kernel<<<(BATCH * HDIM + 255) / 256, 256, 0, stream>>>(Cst, Hbf, s_prev, start_id);

    for (int t = 0; t < T; ++t) {
        prepare_A_kernel<<<(BATCH * KG) / 256, 256, 0, stream>>>(z, emb, s_prev, Hbf, A);
        gates_cell_kernel<<<16, 128, 0, stream>>>(A, Wg, bias, Cst, Hbf);
        logits_kernel<<<VOCAB / 16 / 8, 256, 0, stream>>>(Hbf, Wout, b_out, logits);
        softmax_sample_kernel<<<BATCH, 256, 0, stream>>>(logits, out, s_prev, t, T);
    }
}
